// ScoreNetworkP_54107997995737
// MI455X (gfx1250) — matrix operand order repaired and checked by reading, not compiled, not run on hardware
//
#include <hip/hip_runtime.h>

// ---------------------------------------------------------------------------
// CDNA5 (gfx1250) fused EGNN score network.
// wave32, WMMA bf16 16x16x32 for all 128-deep GEMMs.
// Weights are converted + fragment-swizzled ONCE into a bf16 workspace; the
// hot kernels stage them into LDS with GLOBAL_LOAD_ASYNC_TO_LDS (ASYNCcnt).
// Every B fragment (and k_gemm128's A) is one contiguous 32B LDS vector load.
// ---------------------------------------------------------------------------

typedef __attribute__((ext_vector_type(16))) __bf16 v16bf;
typedef __attribute__((ext_vector_type(8)))  float  v8f;
typedef __attribute__((ext_vector_type(4)))  int    v4i;

#define HID 128
#define NN  96
#define BB  8
#define BN  (BB*NN)      // 768 nodes total
#define JTILES 6         // 96/16
#define NORM_INV 0.01f   // 1/NORM_FACTOR
#define CRANGE 7.5f      // COORDS_RANGE = 15/NBLOCKS

__device__ __forceinline__ float sil (float x){ return x / (1.f + __expf(-x)); }
__device__ __forceinline__ float sigm(float x){ return 1.f / (1.f + __expf(-x)); }
__device__ __forceinline__ float eluf(float x){ return x > 0.f ? x : (__expf(x) - 1.f); }

__device__ __forceinline__ v8f wmma_bf16(v16bf a, v16bf b, v8f c){
  return __builtin_amdgcn_wmma_f32_16x16x32_bf16(false, a, false, b, (short)0, c,
                                                 false, false);
}
__device__ __forceinline__ v8f zero8(){
  v8f z = {0.f,0.f,0.f,0.f,0.f,0.f,0.f,0.f};
  return z;
}

// ------------------------- async global->LDS copy --------------------------
// 16-byte chunk copy using the CDNA5 async-LDS DMA path (ASYNCcnt) when the
// toolchain exposes it; falls back to a plain vector copy otherwise.
__device__ __forceinline__ void async_copy16(const void* g, void* l){
#if __has_builtin(__builtin_amdgcn_global_load_async_to_lds_b128)
  __builtin_amdgcn_global_load_async_to_lds_b128(
      (__attribute__((address_space(1))) v4i*)g,
      (__attribute__((address_space(3))) v4i*)l, 0, 0);
#else
  *(float4*)l = *(const float4*)g;
#endif
}
__device__ __forceinline__ void async_copy_wait(){
#if __has_builtin(__builtin_amdgcn_s_wait_asynccnt)
  __builtin_amdgcn_s_wait_asynccnt(0);
#elif __has_builtin(__builtin_amdgcn_global_load_async_to_lds_b128)
  asm volatile("s_wait_asynccnt 0" ::: "memory");
#endif
}

// --------------------------- fragment swizzles -----------------------------
// B fragment (32x16 of row-major W[k][n]): frag (s,nt), lane = ((kr>>3)&1)*16+(n&15),
// elem e = (kr>>4)*8+(kr&7).  (frag,lane) owns 16 contiguous bf16 (32 B).
__device__ __forceinline__ int bswz(int k, int n){
  int s  = k >> 5, kr = k & 31;
  int lane = ((kr >> 3) & 1)*16 + (n & 15);
  int e  = (kr >> 4)*8 + (kr & 7);
  int nt = n >> 4;
  return ((s*8 + nt)*32 + lane)*16 + e;
}
__device__ __forceinline__ v16bf load_bfragS(const __bf16* WsF, int nt, int s, int lane){
  return *(const v16bf*)(WsF + ((s*8 + nt)*32 + lane)*16);
}

// A fragment (16x32 of row-major X[m][k])
__device__ __forceinline__ int aswz(int m, int k){
  int s  = k >> 5, kr = k & 31;
  int part = kr >> 3;                    // 0..3
  int lane = (part & 1)*16 + m;
  int e    = (part >> 1)*8 + (kr & 7);
  return (s*32 + lane)*16 + e;
}
__device__ __forceinline__ v16bf load_afragS(const __bf16* XsF, int s, int lane){
  return *(const v16bf*)(XsF + (s*32 + lane)*16);
}

// one-time weight convert + swizzle: f32 [128][128] -> bf16 fragment order
__global__ void k_prep_w(const float* __restrict__ W, __bf16* __restrict__ Wo)
{
  int i = blockIdx.x*256 + threadIdx.x;
  if (i >= HID*HID) return;
  Wo[bswz(i >> 7, i & 127)] = (__bf16)W[i];
}

// ---------------------------------------------------------------------------
// Generic fused GEMM: Y[768,128] = epi( X@Wa (+ X2@Wb) + bias ),
// epi = [silu] -> [+residual] -> [*row-mask].  Weights arrive pre-swizzled.
// ---------------------------------------------------------------------------
__global__ __launch_bounds__(256)
void k_gemm128(const float* __restrict__ X,  const __bf16* __restrict__ WaS,
               const float* __restrict__ X2, const __bf16* __restrict__ WbS,
               const float* __restrict__ bias, const float* __restrict__ R,
               const float* __restrict__ mask, float* __restrict__ Y, int act)
{
  __shared__ __attribute__((aligned(32))) __bf16 WsF[HID*HID];   // 32 KB
  __shared__ __attribute__((aligned(32))) __bf16 XsF[16*HID];    // 4 KB
  __shared__ __attribute__((aligned(32))) __bf16 X2sF[16*HID];   // 4 KB
  const int t = threadIdx.x, lane = t & 31, wave = t >> 5;
  const int row0 = blockIdx.x * 16;

  for (int i = t; i < 2048; i += 256)               // 32 KB weight, async DMA
    async_copy16(WaS + i*8, WsF + i*8);
  for (int i = t; i < 16*HID; i += 256)             // A tile: convert+swizzle
    XsF[aswz(i >> 7, i & 127)] = (__bf16)X[row0*HID + i];
  if (X2)
    for (int i = t; i < 16*HID; i += 256)
      X2sF[aswz(i >> 7, i & 127)] = (__bf16)X2[row0*HID + i];
  async_copy_wait();
  __syncthreads();

  const int nt = wave;
  v8f c = zero8();
  #pragma unroll
  for (int s = 0; s < 4; ++s)
    c = wmma_bf16(load_afragS(XsF, s, lane), load_bfragS(WsF, nt, s, lane), c);

  if (X2){
    __syncthreads();
    for (int i = t; i < 2048; i += 256)
      async_copy16(WbS + i*8, WsF + i*8);
    async_copy_wait();
    __syncthreads();
    #pragma unroll
    for (int s = 0; s < 4; ++s)
      c = wmma_bf16(load_afragS(X2sF, s, lane), load_bfragS(WsF, nt, s, lane), c);
  }

  const int n  = nt*16 + (lane & 15);
  const float bn = bias ? bias[n] : 0.f;
  #pragma unroll
  for (int v = 0; v < 8; ++v){
    int m   = v + 8*(lane >> 4);
    int row = row0 + m;
    float y = c[v] + bn;
    if (act) y = sil(y);
    if (R)    y += R[row*HID + n];
    if (mask) y *= mask[row];
    Y[row*HID + n] = y;
  }
}

// ---------------------------------------------------------------------------
// Edge message kernel: per (batch, i-tile of 16 receivers).
// layer1 = silu(Hs[i] + Ht[j] + ea . Wea)  built directly into A fragments,
// layer2 = 128x128 bf16 WMMA, att gating + masked sum over j -> agg/100.
// ---------------------------------------------------------------------------
__global__ __launch_bounds__(256)
void k_edge(const float* __restrict__ Hs,  const float* __restrict__ Ht,
            const float* __restrict__ Wea, // rows 256..258 of edge-W1: [3][128]
            const __bf16* __restrict__ W2S, const float* __restrict__ b2,
            const float* __restrict__ Watt, const float* __restrict__ batt,
            const float* __restrict__ pos,  const float* __restrict__ d0,
            const float* __restrict__ adj,  float* __restrict__ agg)
{
  __shared__ __attribute__((aligned(32))) __bf16 WsF[HID*HID];  // 32 KB
  __shared__ __attribute__((aligned(16))) float  HsS[16*HID];   // 8 KB
  __shared__ __attribute__((aligned(16))) float  HtS[16*HID];   // 8 KB
  __shared__ __attribute__((aligned(16))) float  WeaS[3*HID];
  __shared__ __attribute__((aligned(16))) float  posJS[48];
  __shared__ float  eaS[256*3];
  __shared__ float  posIS[48];
  __shared__ float  b2S[HID], WtS[HID];

  const int t = threadIdx.x, lane = t & 31, wave = t >> 5;
  const int b = blockIdx.x / JTILES, it = blockIdx.x % JTILES, i0 = it*16;
  const int rbase = b*NN + i0;

  for (int i = t; i < 2048; i += 256)               // weight: async DMA
    async_copy16(W2S + i*8, WsF + i*8);
  for (int i = t; i < 512; i += 256)                // Hs i-tile: async DMA
    async_copy16(Hs + rbase*HID + i*4, HsS + i*4);
  if (t < 96) async_copy16(Wea + t*4, WeaS + t*4);
  if (t < 48)  posIS[t] = pos[rbase*3 + t];
  if (t < HID){ b2S[t] = b2[t]; WtS[t] = Watt[t]; }
  const float batt0 = batt[0];

  float aggA[2][8] = {};

  for (int jt = 0; jt < JTILES; ++jt){
    __syncthreads();
    const int j0 = jt*16, jb = b*NN + j0;
    for (int i = t; i < 512; i += 256)              // Ht j-tile: async DMA
      async_copy16(Ht + jb*HID + i*4, HtS + i*4);
    if (t < 12) async_copy16(pos + jb*3 + t*4, posJS + t*4);
    async_copy_wait();
    __syncthreads();
    {
      int ii = t >> 4, jj = t & 15;
      float dx = posIS[ii*3+0]-posJS[jj*3+0];
      float dy = posIS[ii*3+1]-posJS[jj*3+1];
      float dz = posIS[ii*3+2]-posJS[jj*3+2];
      int e = ii*16 + jj;
      int gi = i0+ii, gj = j0+jj;
      eaS[e*3+0] = dx*dx + dy*dy + dz*dz;            // block-level radial
      eaS[e*3+1] = d0 [(b*NN+gi)*NN + gj];           // top-level radial
      eaS[e*3+2] = adj[(b*NN+gi)*NN + gj];
    }
    __syncthreads();

    #pragma unroll
    for (int mt = 0; mt < 2; ++mt){
      const int ii = wave*2 + mt, gi = i0 + ii;
      const int m  = lane & 15;                      // edge row = jj
      const int kb = (lane >> 4) * 8;
      const float e0 = eaS[(ii*16+m)*3+0];
      const float e1 = eaS[(ii*16+m)*3+1];
      const float e2 = eaS[(ii*16+m)*3+2];

      v16bf a[4];                                    // layer-1 in registers
      #pragma unroll
      for (int s = 0; s < 4; ++s){
        #pragma unroll
        for (int e = 0; e < 8; ++e){
          int k = s*32 + kb + e;
          float x = HsS[ii*HID+k] + HtS[m*HID+k]
                  + e0*WeaS[k] + e1*WeaS[HID+k] + e2*WeaS[2*HID+k];
          a[s][e] = (__bf16)sil(x);
          k += 16;
          x = HsS[ii*HID+k] + HtS[m*HID+k]
            + e0*WeaS[k] + e1*WeaS[HID+k] + e2*WeaS[2*HID+k];
          a[s][8+e] = (__bf16)sil(x);
        }
      }

      v8f c[8];
      #pragma unroll
      for (int nt = 0; nt < 8; ++nt){
        c[nt] = zero8();
        #pragma unroll
        for (int s = 0; s < 4; ++s)
          c[nt] = wmma_bf16(a[s], load_bfragS(WsF, nt, s, lane), c[nt]);
      }

      // attention dot over 128 features (cross-lane within 16-groups)
      float attA[8] = {};
      #pragma unroll
      for (int nt = 0; nt < 8; ++nt){
        int n = nt*16 + (lane & 15);
        float bb = b2S[n], wa = WtS[n];
        #pragma unroll
        for (int v = 0; v < 8; ++v){
          float mv = sil(c[nt][v] + bb);
          c[nt][v] = mv;
          attA[v] += mv * wa;
        }
      }
      #pragma unroll
      for (int v = 0; v < 8; ++v){
        attA[v] += __shfl_xor(attA[v], 1, 32);
        attA[v] += __shfl_xor(attA[v], 2, 32);
        attA[v] += __shfl_xor(attA[v], 4, 32);
        attA[v] += __shfl_xor(attA[v], 8, 32);
        int rowm = v + 8*(lane >> 4);
        float g = (gi == j0 + rowm) ? 0.f : 1.f;     // edge mask (no self-loops)
        attA[v] = g * sigm(attA[v] + batt0);
      }
      #pragma unroll
      for (int nt = 0; nt < 8; ++nt){
        float s = 0.f;
        #pragma unroll
        for (int v = 0; v < 8; ++v) s += c[nt][v] * attA[v];
        s += __shfl_xor(s, 16, 32);                  // both halves of j
        aggA[mt][nt] += s;
      }
    }
  }

  if (lane < 16){
    #pragma unroll
    for (int mt = 0; mt < 2; ++mt){
      int gi = i0 + wave*2 + mt;
      #pragma unroll
      for (int nt = 0; nt < 8; ++nt)
        agg[(b*NN+gi)*HID + nt*16 + lane] = aggA[mt][nt] * NORM_INV;
    }
  }
}

// ---------------------------------------------------------------------------
// Coordinate update kernel: same skeleton; epilogue is the 128->1 phi dot,
// tanh(phi)*CRANGE * cdiff accumulated into the position delta.
// ---------------------------------------------------------------------------
__global__ __launch_bounds__(256)
void k_coord(const float* __restrict__ Cs,  const float* __restrict__ Ct,
             const float* __restrict__ Wcea, const __bf16* __restrict__ W2S,
             const float* __restrict__ b2,   const float* __restrict__ Wphi,
             const float* __restrict__ pos,  const float* __restrict__ d0,
             const float* __restrict__ adj,  float* __restrict__ posOut)
{
  __shared__ __attribute__((aligned(32))) __bf16 WsF[HID*HID];
  __shared__ __attribute__((aligned(16))) float  CsS[16*HID];
  __shared__ __attribute__((aligned(16))) float  CtS[16*HID];
  __shared__ __attribute__((aligned(16))) float  WeaS[3*HID];
  __shared__ __attribute__((aligned(16))) float  posJS[48];
  __shared__ float  eaS[256*3];
  __shared__ float  posIS[48];
  __shared__ float  b2S[HID], WpS[HID];

  const int t = threadIdx.x, lane = t & 31, wave = t >> 5;
  const int b = blockIdx.x / JTILES, it = blockIdx.x % JTILES, i0 = it*16;
  const int rbase = b*NN + i0;

  for (int i = t; i < 2048; i += 256)
    async_copy16(W2S + i*8, WsF + i*8);
  for (int i = t; i < 512; i += 256)
    async_copy16(Cs + rbase*HID + i*4, CsS + i*4);
  if (t < 96) async_copy16(Wcea + t*4, WeaS + t*4);
  if (t < 48)  posIS[t] = pos[rbase*3 + t];
  if (t < HID){ b2S[t] = b2[t]; WpS[t] = Wphi[t]; }

  float pAcc[2][3] = {};

  for (int jt = 0; jt < JTILES; ++jt){
    __syncthreads();
    const int j0 = jt*16, jb = b*NN + j0;
    for (int i = t; i < 512; i += 256)
      async_copy16(Ct + jb*HID + i*4, CtS + i*4);
    if (t < 12) async_copy16(pos + jb*3 + t*4, posJS + t*4);
    async_copy_wait();
    __syncthreads();
    {
      int ii = t >> 4, jj = t & 15;
      float dx = posIS[ii*3+0]-posJS[jj*3+0];
      float dy = posIS[ii*3+1]-posJS[jj*3+1];
      float dz = posIS[ii*3+2]-posJS[jj*3+2];
      int e = ii*16 + jj;
      int gi = i0+ii, gj = j0+jj;
      eaS[e*3+0] = dx*dx + dy*dy + dz*dz;
      eaS[e*3+1] = d0 [(b*NN+gi)*NN + gj];
      eaS[e*3+2] = adj[(b*NN+gi)*NN + gj];
    }
    __syncthreads();

    #pragma unroll
    for (int mt = 0; mt < 2; ++mt){
      const int ii = wave*2 + mt, gi = i0 + ii;
      const int m  = lane & 15;
      const int kb = (lane >> 4) * 8;
      const float e0 = eaS[(ii*16+m)*3+0];
      const float e1 = eaS[(ii*16+m)*3+1];
      const float e2 = eaS[(ii*16+m)*3+2];

      v16bf a[4];
      #pragma unroll
      for (int s = 0; s < 4; ++s){
        #pragma unroll
        for (int e = 0; e < 8; ++e){
          int k = s*32 + kb + e;
          float x = CsS[ii*HID+k] + CtS[m*HID+k]
                  + e0*WeaS[k] + e1*WeaS[HID+k] + e2*WeaS[2*HID+k];
          a[s][e] = (__bf16)sil(x);
          k += 16;
          x = CsS[ii*HID+k] + CtS[m*HID+k]
            + e0*WeaS[k] + e1*WeaS[HID+k] + e2*WeaS[2*HID+k];
          a[s][8+e] = (__bf16)sil(x);
        }
      }

      float phiA[8] = {};
      #pragma unroll
      for (int nt = 0; nt < 8; ++nt){
        v8f c = zero8();
        #pragma unroll
        for (int s = 0; s < 4; ++s)
          c = wmma_bf16(a[s], load_bfragS(WsF, nt, s, lane), c);
        int n = nt*16 + (lane & 15);
        float bb = b2S[n], wp = WpS[n];
        #pragma unroll
        for (int v = 0; v < 8; ++v)
          phiA[v] += sil(c[v] + bb) * wp;
      }
      #pragma unroll
      for (int v = 0; v < 8; ++v){
        phiA[v] += __shfl_xor(phiA[v], 1, 32);
        phiA[v] += __shfl_xor(phiA[v], 2, 32);
        phiA[v] += __shfl_xor(phiA[v], 4, 32);
        phiA[v] += __shfl_xor(phiA[v], 8, 32);
        int rowm = v + 8*(lane >> 4);
        float g  = (gi == j0 + rowm) ? 0.f : 1.f;
        float tr = tanhf(phiA[v]) * CRANGE * g;
        float radial = eaS[(ii*16+rowm)*3+0];
        float inv = 1.f / (sqrtf(radial + 1e-8f) + 1.0f);  // NORM_CONST = 1
        float w = tr * inv;
        pAcc[mt][0] += (posIS[ii*3+0] - posJS[rowm*3+0]) * w;
        pAcc[mt][1] += (posIS[ii*3+1] - posJS[rowm*3+1]) * w;
        pAcc[mt][2] += (posIS[ii*3+2] - posJS[rowm*3+2]) * w;
      }
    }
  }

  #pragma unroll
  for (int mt = 0; mt < 2; ++mt)
    #pragma unroll
    for (int cc = 0; cc < 3; ++cc)
      pAcc[mt][cc] += __shfl_xor(pAcc[mt][cc], 16, 32);

  if (lane == 0){
    #pragma unroll
    for (int mt = 0; mt < 2; ++mt){
      int gi = i0 + wave*2 + mt;
      #pragma unroll
      for (int cc = 0; cc < 3; ++cc)
        posOut[(b*NN+gi)*3 + cc] = pos[(b*NN+gi)*3 + cc] + pAcc[mt][cc] * NORM_INV;
    }
  }
}

// ---------------------------------------------------------------------------
// Small VALU kernels
// ---------------------------------------------------------------------------
__global__ void k_embed0(const float* __restrict__ x, const float* __restrict__ tt,
                         const float* __restrict__ W, const float* __restrict__ bias,
                         float* __restrict__ h)
{
  int idx = blockIdx.x*256 + threadIdx.x;      // 768*128
  if (idx >= BN*HID) return;
  int f = idx & (HID-1), row = idx >> 7, b = row / NN;
  const float* xr = x + row*10;
  float y = bias[f];
  #pragma unroll
  for (int k = 0; k < 10; ++k) y += xr[k] * W[k*HID + f];
  y += tt[b] * W[10*HID + f];
  h[idx] = y;
}

__global__ void k_d0(const float* __restrict__ pos, float* __restrict__ d0)
{
  int idx = blockIdx.x*256 + threadIdx.x;      // 8*96*96
  if (idx >= BB*NN*NN) return;
  int j = idx % NN, i = (idx / NN) % NN, b = idx / (NN*NN);
  const float* pi = pos + (b*NN+i)*3;
  const float* pj = pos + (b*NN+j)*3;
  float dx = pi[0]-pj[0], dy = pi[1]-pj[1], dz = pi[2]-pj[2];
  d0[idx] = dx*dx + dy*dy + dz*dz;
}

__global__ void k_pos_slot0(const float* __restrict__ pos, float* __restrict__ plist)
{
  int idx = blockIdx.x*256 + threadIdx.x;      // 768*3
  if (idx >= BN*3) return;
  plist[(idx/3)*9 + (idx%3)] = pos[idx];
}

// delta = (posNew - posIn)*mask ; remove masked mean ; tanh ; store
__global__ void k_posfin(const float* __restrict__ posIn, const float* __restrict__ posNew,
                         const float* __restrict__ flags, float* __restrict__ plist,
                         int slot, float* __restrict__ posOut)
{
  __shared__ float ssum[3];
  __shared__ float scnt;
  int b = blockIdx.x, n = threadIdx.x;
  if (n == 0){ ssum[0]=ssum[1]=ssum[2]=0.f; scnt = 0.f; }
  __syncthreads();
  float d[3] = {0.f,0.f,0.f};
  float fl = 0.f;
  if (n < NN){
    fl = flags[b*NN + n];
    #pragma unroll
    for (int c = 0; c < 3; ++c)
      d[c] = (posNew[(b*NN+n)*3+c] - posIn[(b*NN+n)*3+c]) * fl;
    atomicAdd(&ssum[0], d[0]);
    atomicAdd(&ssum[1], d[1]);
    atomicAdd(&ssum[2], d[2]);
    atomicAdd(&scnt, fl);
  }
  __syncthreads();
  if (n < NN){
    float inv = 1.f / scnt;
    #pragma unroll
    for (int c = 0; c < 3; ++c){
      float v = tanhf((d[c] - ssum[c]*inv) * fl);
      posOut[(b*NN+n)*3 + c]          = v;
      plist [(b*NN+n)*9 + slot*3 + c] = v;
    }
  }
}

// final head: 9 -> 18 -> 18 -> 3 with ELU, masked
__global__ void k_final(const float* __restrict__ plist, const float* __restrict__ flags,
                        const float* __restrict__ W0, const float* __restrict__ b0,
                        const float* __restrict__ W1, const float* __restrict__ b1,
                        const float* __restrict__ W2, const float* __restrict__ b2,
                        float* __restrict__ out)
{
  int row = blockIdx.x*128 + threadIdx.x;
  if (row >= BN) return;
  float pv[9];
  #pragma unroll
  for (int k = 0; k < 9; ++k) pv[k] = plist[row*9 + k];
  float h1[18];
  #pragma unroll
  for (int o = 0; o < 18; ++o){
    float y = b0[o];
    #pragma unroll
    for (int k = 0; k < 9; ++k) y += pv[k] * W0[k*18 + o];
    h1[o] = eluf(y);
  }
  float h2[18];
  #pragma unroll
  for (int o = 0; o < 18; ++o){
    float y = b1[o];
    #pragma unroll
    for (int k = 0; k < 18; ++k) y += h1[k] * W1[k*18 + o];
    h2[o] = eluf(y);
  }
  float fl = flags[row];
  #pragma unroll
  for (int o = 0; o < 3; ++o){
    float y = b2[o];
    #pragma unroll
    for (int k = 0; k < 18; ++k) y += h2[k] * W2[k*18 + o];
    out[row*3 + o] = y * fl;
  }
}

// ---------------------------------------------------------------------------
// Host orchestration
// ---------------------------------------------------------------------------
struct BlkP {
  const float *Wc0,*bc0,*Wc1,*bc1,*Wc2;          // coord MLP (last has no bias)
  const float *Wa,*ba;                           // attention
  const float *We0,*be0,*We1,*be1;               // edge MLP
  const float *Wn0,*bn0,*Wn1,*bn1;               // node MLP
};
struct LayP { BlkP blk[2]; const float *Wemb,*bemb,*Wout,*bout; };
struct BlkS {
  const __bf16 *We0lo,*We0hi,*We1,*Wn0lo,*Wn0hi,*Wn1,*Wc0lo,*Wc0hi,*Wc1;
};
struct LayS { BlkS blk[2]; const __bf16 *Wemb,*Wout; };

extern "C" void kernel_launch(void* const* d_in, const int* in_sizes, int n_in,
                              void* d_out, int out_size, void* d_ws, size_t ws_size,
                              hipStream_t stream)
{
  (void)in_sizes; (void)n_in; (void)out_size; (void)ws_size;
  const float* x     = (const float*)d_in[0];
  const float* pos   = (const float*)d_in[1];
  const float* adj   = (const float*)d_in[2];
  const float* flags = (const float*)d_in[3];
  const float* tt    = (const float*)d_in[4];

  // params flattened in jax pytree order (dict keys sorted): "final" then "layers"
  int pi = 5;
  auto nxt = [&](){ return (const float*)d_in[pi++]; };
  const float *fW0=nxt(),*fb0=nxt(),*fW1=nxt(),*fb1=nxt(),*fW2=nxt(),*fb2=nxt();
  LayP L[2];
  for (int l = 0; l < 2; ++l){
    for (int k = 0; k < 2; ++k){        // block: "coord" then "gcls"
      BlkP& B = L[l].blk[k];
      B.Wc0=nxt(); B.bc0=nxt(); B.Wc1=nxt(); B.bc1=nxt(); B.Wc2=nxt();
      B.Wa =nxt(); B.ba =nxt();          // gcl: "att","edge","node"
      B.We0=nxt(); B.be0=nxt(); B.We1=nxt(); B.be1=nxt();
      B.Wn0=nxt(); B.bn0=nxt(); B.Wn1=nxt(); B.bn1=nxt();
    }
    L[l].Wemb=nxt(); L[l].bemb=nxt(); L[l].Wout=nxt(); L[l].bout=nxt();
  }

  // workspace (floats)
  float* w = (float*)d_ws;
  float* hA   = w; w += BN*HID;
  float* hB   = w; w += BN*HID;
  float* Hsb  = w; w += BN*HID;
  float* Htb  = w; w += BN*HID;
  float* aggb = w; w += BN*HID;
  float* Tb   = w; w += BN*HID;
  float* d0b  = w; w += BB*NN*NN;
  float* pB0  = w; w += BN*3;
  float* pB1  = w; w += BN*3;
  float* pB2  = w; w += BN*3;
  float* plist= w; w += BN*9;
  __bf16* swz = (__bf16*)w; w += 39*8192;      // 39 swizzled 128x128 bf16 mats

  const dim3 T256(256), T128(128);
  const int GEMM_G = BN/16;            // 48
  const int EDGE_G = BB*JTILES;        // 48

  // ---- one-time weight convert + swizzle (stream-ordered, graph-safe) ----
  int swzCnt = 0;
  auto prep = [&](const float* W)->const __bf16* {
    __bf16* dst = swz + (size_t)swzCnt * (HID*HID);
    ++swzCnt;
    k_prep_w<<<(HID*HID+255)/256, T256, 0, stream>>>(W, dst);
    return dst;
  };
  LayS S[2];
  for (int l = 0; l < 2; ++l){
    for (int k = 0; k < 2; ++k){
      const BlkP& B = L[l].blk[k];
      BlkS& Z = S[l].blk[k];
      Z.We0lo = prep(B.We0);
      Z.We0hi = prep(B.We0 + 128*HID);
      Z.We1   = prep(B.We1);
      Z.Wn0lo = prep(B.Wn0);
      Z.Wn0hi = prep(B.Wn0 + 128*HID);
      Z.Wn1   = prep(B.Wn1);
      Z.Wc0lo = prep(B.Wc0);
      Z.Wc0hi = prep(B.Wc0 + 128*HID);
      Z.Wc1   = prep(B.Wc1);
    }
    S[l].Wout = prep(L[l].Wout);
    S[l].Wemb = (l > 0) ? prep(L[l].Wemb) : nullptr;
  }

  k_pos_slot0<<<(BN*3+255)/256, T256, 0, stream>>>(pos, plist);

  float* h  = hA;
  float* hN = hB;
  const float* posCur = pos;           // layer-0 input positions (read-only)
  float* scratch[3] = {pB0, pB1, pB2};

  for (int l = 0; l < 2; ++l){
    k_d0<<<(BB*NN*NN+255)/256, T256, 0, stream>>>(posCur, d0b);

    if (l == 0){
      k_embed0<<<(BN*HID+255)/256, T256, 0, stream>>>(x, tt, L[0].Wemb, L[0].bemb, h);
    } else {
      k_gemm128<<<GEMM_G, T256, 0, stream>>>(h, S[l].Wemb, nullptr, nullptr,
                                             L[l].bemb, nullptr, nullptr, hN, 0);
      float* tmp = h; h = hN; hN = tmp;
    }

    const float* posLayerIn = posCur;
    float* pick[3]; int npick = 0;
    for (int i = 0; i < 3; ++i)
      if (scratch[i] != posCur) pick[npick++] = scratch[i];

    for (int k = 0; k < 2; ++k){
      const BlkP& B = L[l].blk[k];
      const BlkS& Z = S[l].blk[k];
      // edge-MLP layer-1 factorized: Hs = h@W1[src], Ht = h@W1[tgt] + b1
      k_gemm128<<<GEMM_G, T256, 0, stream>>>(h, Z.We0lo, nullptr, nullptr,
                                             nullptr, nullptr, nullptr, Hsb, 0);
      k_gemm128<<<GEMM_G, T256, 0, stream>>>(h, Z.We0hi, nullptr, nullptr,
                                             B.be0, nullptr, nullptr, Htb, 0);
      k_edge<<<EDGE_G, T256, 0, stream>>>(Hsb, Htb, B.We0 + 256*HID,
                                          Z.We1, B.be1, B.Wa, B.ba,
                                          posCur, d0b, adj, aggb);
      // node MLP: T = silu(h@Wn0[:128] + agg@Wn0[128:] + bn0); h = (h + T@Wn1 + bn1)*mask
      k_gemm128<<<GEMM_G, T256, 0, stream>>>(h, Z.Wn0lo, aggb, Z.Wn0hi,
                                             B.bn0, nullptr, nullptr, Tb, 1);
      k_gemm128<<<GEMM_G, T256, 0, stream>>>(Tb, Z.Wn1, nullptr, nullptr,
                                             B.bn1, h, flags, hN, 0);
      { float* tmp = h; h = hN; hN = tmp; }
      // coord-MLP layer-1 factorized on updated h
      k_gemm128<<<GEMM_G, T256, 0, stream>>>(h, Z.Wc0lo, nullptr, nullptr,
                                             nullptr, nullptr, nullptr, Hsb, 0);
      k_gemm128<<<GEMM_G, T256, 0, stream>>>(h, Z.Wc0hi, nullptr, nullptr,
                                             B.bc0, nullptr, nullptr, Htb, 0);
      float* posNext = pick[k];
      k_coord<<<EDGE_G, T256, 0, stream>>>(Hsb, Htb, B.Wc0 + 256*HID,
                                           Z.Wc1, B.bc1, B.Wc2,
                                           posCur, d0b, adj, posNext);
      posCur = posNext;
    }

    k_gemm128<<<GEMM_G, T256, 0, stream>>>(h, S[l].Wout, nullptr, nullptr,
                                           L[l].bout, nullptr, flags, hN, 0);
    { float* tmp = h; h = hN; hN = tmp; }

    // delta, masked mean removal, tanh -> next layer positions + pos_list slot
    k_posfin<<<BB, T128, 0, stream>>>(posLayerIn, posCur, flags, plist, l+1, pick[0]);
    posCur = pick[0];
  }

  k_final<<<(BN+127)/128, T128, 0, stream>>>(plist, flags, fW0, fb0, fW1, fb1,
                                             fW2, fb2, (float*)d_out);
}
